// Decoder_91036126806992
// MI455X (gfx1250) — matrix operand order repaired and checked by reading, not compiled, not run on hardware
//
#include <hip/hip_runtime.h>
#include <hip/hip_bf16.h>

// ---------------------------------------------------------------------------
// LAS attention decoder for MI455X (gfx1250, wave32).
//
// B=128, T_ENC=1024, KVS=128, EMBED=256, VOCAB=30, MAX_LEN=250.
//
//  * 32 persistent blocks x 512 threads (16 waves). Block g owns 4 batches.
//    All per-batch state is block-private -> no cross-block sync; the whole
//    250-step recurrence runs inside one kernel. h/c/ctx live in LDS.
//  * Dominant cost: streaming key+value every step. Converted to bf16 in
//    workspace (64MB, L2-resident in the 192MB L2) -> 16GB of L2 reads over
//    the decode, spread across 32 WGPs for aggregate L2 bandwidth.
//    (fp32 fallback path if ws_size is too small.)
//  * Gate GEMM via v_wmma_f32_16x16x32_bf16 (fp32 accum). A laundered int
//    offset (not a laundered pointer: keeps global addrspace inference ->
//    global_load, not flat_load) stops the compiler hoisting 256 VGPRs of
//    weight fragments out of the t-loop. One A-fragment load feeds two
//    WMMAs (two N-tiles per wave).
//  * Attention: 4 waves per batch, each owning a 256-wide T-chunk; softmax
//    max/sum combined through a small LDS array; context partials combined
//    with a deterministic tree sum (no FP atomics).
// ---------------------------------------------------------------------------

typedef __attribute__((ext_vector_type(16))) __bf16 v16bf;
typedef __attribute__((ext_vector_type(8)))  __bf16 v8bf;
typedef __attribute__((ext_vector_type(8)))  float  v8f;

#define N_B      128
#define T_ENC    1024
#define KVS      128
#define EMBED    256
#define VOCAB    30
#define MAX_LEN  250

#define TPB      512          // 16 waves of 32
#define BPB      4            // batches per block
#define NBLK     (N_B / BPB)  // 32 blocks
#define LDX      392          // padded X row (384 + 8), keeps 16B alignment
#define PRED_OFF ((size_t)N_B * MAX_LEN * VOCAB)   // 960000 floats

#define KV_ELEMS ((size_t)N_B * T_ENC * KVS)       // 16,777,216 per tensor
#define WS_WBYTES ((size_t)(512 * 384 + 512 * 128) * 2)   // 524288

// ---------------------------------------------------------------------------
// Init kernels
// ---------------------------------------------------------------------------
__global__ void cvt_weights_kernel(const float* __restrict__ Wih,
                                   const float* __restrict__ Whh,
                                   __bf16* __restrict__ oWih,
                                   __bf16* __restrict__ oWhh) {
    int i = blockIdx.x * blockDim.x + threadIdx.x;
    if (i < 512 * 384) oWih[i] = (__bf16)Wih[i];
    if (i < 512 * 128) oWhh[i] = (__bf16)Whh[i];
}

__global__ void cvt_kv_kernel(const float* __restrict__ k,
                              const float* __restrict__ v,
                              __bf16* __restrict__ ok,
                              __bf16* __restrict__ ov) {
    size_t i = (size_t)blockIdx.x * blockDim.x + threadIdx.x;
    if (i < KV_ELEMS) {
        ok[i] = (__bf16)k[i];
        ov[i] = (__bf16)v[i];
    }
}

// exact bf16(lo)/bf16(hi) of a packed dword -> f32
__device__ __forceinline__ float bf16lo(unsigned u) {
    return __uint_as_float(u << 16);
}
__device__ __forceinline__ float bf16hi(unsigned u) {
    return __uint_as_float(u & 0xffff0000u);
}

// ---------------------------------------------------------------------------
// Main persistent decoder kernel (templated on K/V element type).
// ---------------------------------------------------------------------------
template <bool KVBF>
__global__ __launch_bounds__(TPB, 1)
void decoder_kernel(const void*  __restrict__ keykv,
                    const void*  __restrict__ valkv,
                    const float* __restrict__ value_f32,   // for ctx0 init
                    const int*   __restrict__ y,
                    const float* __restrict__ emb,
                    const float* __restrict__ b_ih,
                    const float* __restrict__ b_hh,
                    const float* __restrict__ b_cp,
                    const __bf16* __restrict__ Wih,   // [512][384] bf16
                    const __bf16* __restrict__ Whh,   // [512][128] bf16
                    float* __restrict__ out) {
    extern __shared__ char smemRaw[];

    // ---- LDS partition (~80 KB) ----
    __bf16* XB = (__bf16*)smemRaw;            // [16][LDX]  bf16 activations
    __bf16* HB = XB + 16 * LDX;               // [16][128]  bf16 hidden
    float*  G  = (float*)(HB + 16 * KVS);     // [16][512]  fp32 gates
    float*  HS = G  + 16 * 512;               // [BPB][128] h
    float*  CS = HS + BPB * KVS;              // [BPB][128] c
    float*  CX = CS + BPB * KVS;              // [BPB][128] ctx
    float*  EN = CX + BPB * KVS;              // [BPB][1024] energy/attn
    float*  PC = EN + BPB * T_ENC;            // [4][BPB][128] ctx partials
    float*  RED = PC + 4 * BPB * KVS;         // [BPB][8] softmax reduce

    const int tid  = threadIdx.x;
    const int lane = tid & 31;
    const int wave = tid >> 5;                // 0..15
    const int ab   = wave >> 2;               // attention batch 0..3
    const int ap   = wave & 3;                // attention T-chunk 0..3
    const int B0   = blockIdx.x * BPB;

    // ---- init: zero padded activation rows, h=c=0, ctx = value[:,0,:] ----
    for (int i = tid; i < 16 * LDX; i += TPB) XB[i] = (__bf16)0.f;
    for (int i = tid; i < 16 * KVS; i += TPB) HB[i] = (__bf16)0.f;
    for (int i = tid; i < BPB * KVS; i += TPB) {
        int b = i >> 7, k = i & 127;
        HS[i] = 0.f; CS[i] = 0.f;
        CX[i] = value_f32[((size_t)(B0 + b) * T_ENC + 0) * KVS + k];
    }
    __syncthreads();

    for (int t = 0; t < MAX_LEN; ++t) {
        // ---- phase 1: build x = [emb(y[t-1]) | ctx] as bf16 ----
        for (int i = tid; i < BPB * EMBED; i += TPB) {
            int b = i >> 8, e = i & 255;
            float v = 0.f;
            if (t > 0) {
                int tok = y[(B0 + b) * MAX_LEN + (t - 1)];
                v = emb[tok * EMBED + e];
            }
            XB[b * LDX + e] = (__bf16)v;
        }
        for (int i = tid; i < BPB * KVS; i += TPB) {
            int b = i >> 7, k = i & 127;
            XB[b * LDX + EMBED + k] = (__bf16)CX[i];
        }
        __syncthreads();

        // ---- phase 2: gate GEMM via WMMA bf16 (2 N-tiles/wave, shared A) ----
        {
            // Launder an offset (kept 0) so weight-fragment loads cannot be
            // hoisted/CSE'd across t-steps (would spill 256 VGPRs). Keeping
            // the base pointer intact preserves global addrspace inference.
            int lz = 0;
            asm volatile("" : "+s"(lz));
            const __bf16* WihT = Wih + lz;
            const __bf16* WhhT = Whh + lz;

            const int row   = lane & 15;             // A batch row / B out col
            const int aHalf = (lane < 16) ? 0 : 16;  // A K-halves
            const int bHalf = (lane < 16) ? 0 : 16;  // B K-halves
            const int mBase = (lane < 16) ? 0 : 8;   // C/D row base
            const int o0    = wave * 32;             // two tiles: o0, o0+16

            v8f acc0 = {}, acc1 = {};

            const __bf16* wihRow0 = WihT + (size_t)(o0 + row) * 384;
            const __bf16* wihRow1 = WihT + (size_t)(o0 + 16 + row) * 384;
            #pragma unroll
            for (int k0 = 0; k0 < 384; k0 += 32) {
                const __bf16* pa = XB + row * LDX + k0 + aHalf;
                v8bf lo = *(const v8bf*)pa;
                v8bf hi = *(const v8bf*)(pa + 8);
                v16bf a = __builtin_shufflevector(
                    lo, hi, 0,1,2,3,4,5,6,7,8,9,10,11,12,13,14,15);
                v16bf bm0 = *(const v16bf*)(wihRow0 + k0 + bHalf);
                v16bf bm1 = *(const v16bf*)(wihRow1 + k0 + bHalf);
                acc0 = __builtin_amdgcn_wmma_f32_16x16x32_bf16(
                    false, a, false, bm0, (short)0, acc0, false, false);
                acc1 = __builtin_amdgcn_wmma_f32_16x16x32_bf16(
                    false, a, false, bm1, (short)0, acc1, false, false);
            }

            const __bf16* whhRow0 = WhhT + (size_t)(o0 + row) * 128;
            const __bf16* whhRow1 = WhhT + (size_t)(o0 + 16 + row) * 128;
            #pragma unroll
            for (int k0 = 0; k0 < 128; k0 += 32) {
                const __bf16* pa = HB + row * KVS + k0 + aHalf;
                v8bf lo = *(const v8bf*)pa;
                v8bf hi = *(const v8bf*)(pa + 8);
                v16bf a = __builtin_shufflevector(
                    lo, hi, 0,1,2,3,4,5,6,7,8,9,10,11,12,13,14,15);
                v16bf bm0 = *(const v16bf*)(whhRow0 + k0 + bHalf);
                v16bf bm1 = *(const v16bf*)(whhRow1 + k0 + bHalf);
                acc0 = __builtin_amdgcn_wmma_f32_16x16x32_bf16(
                    false, a, false, bm0, (short)0, acc0, false, false);
                acc1 = __builtin_amdgcn_wmma_f32_16x16x32_bf16(
                    false, a, false, bm1, (short)0, acc1, false, false);
            }

            #pragma unroll
            for (int r8 = 0; r8 < 8; ++r8) {
                G[(mBase + r8) * 512 + o0 + row]      = acc0[r8];
                G[(mBase + r8) * 512 + o0 + 16 + row] = acc1[r8];
            }
        }
        __syncthreads();

        // ---- phase 3: LSTM pointwise (BPB*KVS = 512 units, 1/thread) ----
        for (int i = tid; i < BPB * KVS; i += TPB) {
            int b = i >> 7, k = i & 127;
            float gi = G[b * 512 +       k] + b_ih[      k] + b_hh[      k];
            float gf = G[b * 512 + 128 + k] + b_ih[128 + k] + b_hh[128 + k];
            float gg = G[b * 512 + 256 + k] + b_ih[256 + k] + b_hh[256 + k];
            float go = G[b * 512 + 384 + k] + b_ih[384 + k] + b_hh[384 + k];
            float si = 1.f / (1.f + __expf(-gi));
            float sf = 1.f / (1.f + __expf(-gf));
            float so = 1.f / (1.f + __expf(-go));
            float cn = sf * CS[i] + si * tanhf(gg);
            float hn = so * tanhf(cn);
            CS[i] = cn;
            HS[i] = hn;
            HB[i] = (__bf16)hn;
        }
        __syncthreads();

        // ---- phase 4a: energy over this wave's T-chunk; partial max ----
        const int tbeg = ap * (T_ENC / 4);
        const int tend = tbeg + (T_ENC / 4);
        const float* hq = HS + ab * KVS;
        {
            float m = -1e30f;
            for (int tt = tbeg + lane; tt < tend; tt += 32) {
                const size_t rowoff = ((size_t)(B0 + ab) * T_ENC + tt) * KVS;
                float d = 0.f;
                if (KVBF) {
                    const __bf16* kr = (const __bf16*)keykv + rowoff;
                    #pragma unroll
                    for (int k0 = 0; k0 < KVS; k0 += 8) {
                        v8bf kv = *(const v8bf*)(kr + k0);
                        #pragma unroll
                        for (int j = 0; j < 8; ++j)
                            d += (float)kv[j] * hq[k0 + j];
                    }
                } else {
                    const float4* kr = (const float4*)((const float*)keykv + rowoff);
                    const float4* hq4 = (const float4*)hq;
                    #pragma unroll 8
                    for (int k4 = 0; k4 < KVS / 4; ++k4) {
                        float4 kv = kr[k4];
                        float4 hv = hq4[k4];
                        d += kv.x * hv.x + kv.y * hv.y + kv.z * hv.z + kv.w * hv.w;
                    }
                }
                EN[ab * T_ENC + tt] = d;
                m = fmaxf(m, d);
            }
            for (int off = 16; off; off >>= 1) m = fmaxf(m, __shfl_xor(m, off, 32));
            if (lane == 0) RED[ab * 8 + ap] = m;
        }
        __syncthreads();

        // ---- phase 4b: global max, exp over chunk, partial sum ----
        {
            float m = fmaxf(fmaxf(RED[ab * 8 + 0], RED[ab * 8 + 1]),
                            fmaxf(RED[ab * 8 + 2], RED[ab * 8 + 3]));
            float s = 0.f;
            for (int tt = tbeg + lane; tt < tend; tt += 32) {
                float e = __expf(EN[ab * T_ENC + tt] - m);
                EN[ab * T_ENC + tt] = e;
                s += e;
            }
            for (int off = 16; off; off >>= 1) s += __shfl_xor(s, off, 32);
            if (lane == 0) RED[ab * 8 + 4 + ap] = s;
        }
        __syncthreads();

        // ---- phase 4c: normalize chunk; context partials over chunk ----
        {
            float s = (RED[ab * 8 + 4] + RED[ab * 8 + 5]) +
                      (RED[ab * 8 + 6] + RED[ab * 8 + 7]);
            float inv = 1.f / s;

            for (int tt = tbeg + lane; tt < tend; tt += 32) {
                float a = EN[ab * T_ENC + tt] * inv;
                EN[ab * T_ENC + tt] = a;
                if (blockIdx.x == 0 && ab == 0)
                    out[PRED_OFF + (size_t)t * T_ENC + tt] = a;  // attn[0]
            }

            float* pc = PC + (ap * BPB + ab) * KVS;
            if (KVBF) {
                // lane covers k = {2*lane, 2*lane+1, 64+2*lane, 64+2*lane+1}
                float a0 = 0.f, a1 = 0.f, a2 = 0.f, a3 = 0.f;
                for (int tt = tbeg; tt < tend; ++tt) {
                    float a = EN[ab * T_ENC + tt];
                    const size_t rowoff = ((size_t)(B0 + ab) * T_ENC + tt) * KVS;
                    const unsigned* vr =
                        (const unsigned*)((const __bf16*)valkv + rowoff);
                    unsigned u0 = vr[lane];        // k = 2*lane, 2*lane+1
                    unsigned u1 = vr[lane + 32];   // k = 64+2*lane, 64+2*lane+1
                    a0 += a * bf16lo(u0);
                    a1 += a * bf16hi(u0);
                    a2 += a * bf16lo(u1);
                    a3 += a * bf16hi(u1);
                }
                pc[2 * lane]          = a0;
                pc[2 * lane + 1]      = a1;
                pc[64 + 2 * lane]     = a2;
                pc[64 + 2 * lane + 1] = a3;
            } else {
                float a0 = 0.f, a1 = 0.f, a2 = 0.f, a3 = 0.f;
                for (int tt = tbeg; tt < tend; ++tt) {
                    float a = EN[ab * T_ENC + tt];
                    const size_t rowoff = ((size_t)(B0 + ab) * T_ENC + tt) * KVS;
                    const float* vr = (const float*)valkv + rowoff;
                    a0 += a * vr[lane];
                    a1 += a * vr[lane + 32];
                    a2 += a * vr[lane + 64];
                    a3 += a * vr[lane + 96];
                }
                pc[lane]      = a0;
                pc[lane + 32] = a1;
                pc[lane + 64] = a2;
                pc[lane + 96] = a3;
            }
        }
        __syncthreads();

        // ---- phase 4d: deterministic combine of ctx partials ----
        for (int i = tid; i < BPB * KVS; i += TPB) {
            CX[i] = (PC[0 * BPB * KVS + i] + PC[1 * BPB * KVS + i]) +
                    (PC[2 * BPB * KVS + i] + PC[3 * BPB * KVS + i]);
        }
        __syncthreads();

        // ---- phase 5: pred = [h|ctx] @ emb^T + b_cp ----
        for (int i = tid; i < BPB * VOCAB; i += TPB) {
            int b = i / VOCAB, vc = i % VOCAB;
            const float* er = emb + vc * EMBED;
            const float* hh = HS + b * KVS;
            const float* cx = CX + b * KVS;
            float d = b_cp[vc];
            #pragma unroll 4
            for (int e = 0; e < KVS; ++e) d += hh[e] * er[e];
            #pragma unroll 4
            for (int e = 0; e < KVS; ++e) d += cx[e] * er[KVS + e];
            out[((size_t)(B0 + b) * MAX_LEN + t) * VOCAB + vc] = d;
        }
        __syncthreads();
    }
}

// ---------------------------------------------------------------------------
// Host launch
// ---------------------------------------------------------------------------
extern "C" void kernel_launch(void* const* d_in, const int* in_sizes, int n_in,
                              void* d_out, int out_size, void* d_ws, size_t ws_size,
                              hipStream_t stream) {
    const float* key   = (const float*)d_in[0];
    const float* value = (const float*)d_in[1];
    // d_in[2] = encoder_len (unused by the reference computation)
    const int*   y     = (const int*)d_in[3];
    const float* emb   = (const float*)d_in[4];
    const float* W_ih  = (const float*)d_in[5];
    const float* W_hh  = (const float*)d_in[6];
    const float* b_ih  = (const float*)d_in[7];
    const float* b_hh  = (const float*)d_in[8];
    const float* b_cp  = (const float*)d_in[9];
    float* out = (float*)d_out;

    // workspace layout
    __bf16* Wihb = (__bf16*)d_ws;                       // 512*384
    __bf16* Whhb = Wihb + (size_t)512 * 384;            // 512*128
    __bf16* keyb = (__bf16*)((char*)d_ws + WS_WBYTES);  // 16M bf16
    __bf16* valb = keyb + KV_ELEMS;                     // 16M bf16

    const size_t ws_need_kv = WS_WBYTES + 2 * KV_ELEMS * sizeof(__bf16);
    const bool   use_bf16kv = (ws_size >= ws_need_kv);  // constant per run

    cvt_weights_kernel<<<(512 * 384 + 255) / 256, 256, 0, stream>>>(
        W_ih, W_hh, Wihb, Whhb);

    const size_t smem =
        (size_t)16 * LDX * sizeof(__bf16) +        // XB
        (size_t)16 * KVS * sizeof(__bf16) +        // HB
        (size_t)16 * 512 * sizeof(float)  +        // G
        (size_t)BPB * KVS * sizeof(float) * 3 +    // HS, CS, CX
        (size_t)BPB * T_ENC * sizeof(float) +      // EN
        (size_t)4 * BPB * KVS * sizeof(float) +    // PC
        (size_t)BPB * 8 * sizeof(float);           // RED

    static bool attr_set = false;                  // host-side only; deterministic
    if (!attr_set) {
        (void)hipFuncSetAttribute(
            reinterpret_cast<const void*>(&decoder_kernel<true>),
            hipFuncAttributeMaxDynamicSharedMemorySize, (int)smem);
        (void)hipFuncSetAttribute(
            reinterpret_cast<const void*>(&decoder_kernel<false>),
            hipFuncAttributeMaxDynamicSharedMemorySize, (int)smem);
        attr_set = true;
    }

    if (use_bf16kv) {
        cvt_kv_kernel<<<(int)((KV_ELEMS + 255) / 256), 256, 0, stream>>>(
            key, value, keyb, valb);
        decoder_kernel<true><<<NBLK, TPB, smem, stream>>>(
            keyb, valb, value, y, emb, b_ih, b_hh, b_cp, Wihb, Whhb, out);
    } else {
        decoder_kernel<false><<<NBLK, TPB, smem, stream>>>(
            key, value, value, y, emb, b_ih, b_hh, b_cp, Wihb, Whhb, out);
    }
}
